// GATModel_77575699300937
// MI455X (gfx1250) — matrix operand order repaired and checked by reading, not compiled, not run on hardware
//
#include <hip/hip_runtime.h>
#include <math.h>

#define N_NODES 50000
#define NEDGE   800000
#define ETOT    (NEDGE + N_NODES)   // edges + self loops = 850000
#define DIM     128
#define NLAYER  3
#define GGR     64
#define EMBD    64
#define NEG_SLOPE 0.2f

#define XS_PITCH (DIM + 2)   // fp32 x-tile row pitch in LDS (TDM pads 2 dwords / 128)
#define WT_PITCH (DIM + 4)   // bf16 W row pitch in LDS (conflict-free half-wave reads)
#define WT_DWORDS (DIM * WT_PITCH / 2)   // 8448 dwords for the whole pitched W buffer

typedef __attribute__((ext_vector_type(16))) __bf16 v16bf;
typedef __attribute__((ext_vector_type(8)))  float  v8f;
typedef unsigned u32x4 __attribute__((ext_vector_type(4)));
typedef int      i32x4 __attribute__((ext_vector_type(4)));
typedef int      i32x8 __attribute__((ext_vector_type(8)));

#if defined(__has_builtin)
#if __has_builtin(__builtin_amdgcn_tensor_load_to_lds)
#define HAVE_TDM 1
#endif
#if __has_builtin(__builtin_amdgcn_s_wait_tensorcnt)
#define HAVE_TENSORCNT 1
#endif
#endif

// ---------- helpers ----------

__device__ __forceinline__ __bf16 f2bf(float f) {
  union { float f; unsigned u; } v; v.f = f;
  unsigned r = (v.u + 0x7FFFu + ((v.u >> 16) & 1u)) >> 16;  // round-to-nearest-even
  union { unsigned short s; __bf16 b; } o; o.s = (unsigned short)r;
  return o.b;
}

// order-preserving float <-> uint for atomicMax-based segment max
__device__ __forceinline__ unsigned f2ord(float f) {
  unsigned u = __float_as_uint(f);
  return (u & 0x80000000u) ? ~u : (u | 0x80000000u);
}
__device__ __forceinline__ float ord2f(unsigned u) {
  unsigned b = (u & 0x80000000u) ? (u & 0x7FFFFFFFu) : ~u;
  return __uint_as_float(b);
}

__device__ __forceinline__ void edge_sd(const int* __restrict__ ei, int i, int& s, int& d) {
  if (i < NEDGE) { s = ei[i]; d = ei[NEDGE + i]; }
  else           { s = i - NEDGE; d = s; }      // appended self loop
}

#ifdef HAVE_TDM
__device__ __forceinline__ void tdm_wait0() {
#ifdef HAVE_TENSORCNT
  __builtin_amdgcn_s_wait_tensorcnt(0);
#else
  asm volatile("s_wait_tensorcnt 0x0" ::: "memory");
#endif
}
#endif

// ---------- once per layer: W (fp32, row-major [k][n]) -> bf16 transposed+pitched ----------

__global__ __launch_bounds__(256)
void gat_wconv_kernel(const float* __restrict__ W, unsigned short* __restrict__ wt)
{
  int i = blockIdx.x * blockDim.x + threadIdx.x;
  if (i >= DIM * DIM) return;
  int k = i >> 7, n = i & 127;
  union { float f; unsigned u; } v; v.f = W[i];
  unsigned r = (v.u + 0x7FFFu + ((v.u >> 16) & 1u)) >> 16;
  wt[n * WT_PITCH + k] = (unsigned short)r;
}

// ---------- fused GEMM (h = x@W, bf16 WMMA, fp32 acc) + alpha_src/alpha_dst ----------
// Block = 256 threads = 8 waves; each wave owns 16 rows and all 128 output cols.
// Wave 0 TDMs the 128x128 fp32 x tile into LDS (row-pitch padded by the D#),
// wave 1 TDMs the pre-converted bf16 W buffer. OOB rows zero-fill via tensor-dim clamp.

struct GemmShared {
  __bf16 Wt[DIM * WT_PITCH];     // Wt[n * WT_PITCH + k] = bf16(W[k][n]),   ~33 KB
  float  xs[DIM * XS_PITCH];     // xs[r * XS_PITCH + c] = x[r0 + r][c],    ~65 KB
};

__global__ __launch_bounds__(256)
void gat_gemm_kernel(const float* __restrict__ x, const unsigned short* __restrict__ wt,
                     const float* __restrict__ asrc, const float* __restrict__ adst,
                     float* __restrict__ h, float* __restrict__ alpha_s,
                     float* __restrict__ alpha_d)
{
  __shared__ GemmShared sm;

  const int tid = threadIdx.x;
  const int r0  = blockIdx.x * 128;

#ifdef HAVE_TDM
  if (tid < 32) {                       // wave 0: x tile DMA (2D, padded rows)
    unsigned ldsOff = (unsigned)((char*)&sm.xs[0] - (char*)&sm);
    unsigned long long ga = (unsigned long long)(size_t)(x + (size_t)r0 * DIM);
    int rem = N_NODES - r0;
    unsigned td1 = rem > 0 ? (unsigned)rem : 0u;   // rows available -> zero-fill past end

    u32x4 g0;
    g0[0] = 1u;                                          // count=1, normal mode
    g0[1] = ldsOff;                                      // lds_addr
    g0[2] = (unsigned)(ga & 0xFFFFFFFFull);              // global_addr[31:0]
    g0[3] = (unsigned)((ga >> 32) & 0x1FFFFFFull) | (2u << 30);  // addr[56:32] | type=2

    i32x8 g1;
    // data_size=4B(2), pad_enable, pad_interval=128 dwords(6), pad_amount=2 dwords(1)
    g1[0] = (int)((2u << 16) | (1u << 20) | (6u << 22) | (1u << 25));
    g1[1] = (int)(((unsigned)DIM & 0xFFFFu) << 16);      // tensor_dim0 lo
    g1[2] = (int)(((unsigned)DIM >> 16) | ((td1 & 0xFFFFu) << 16));  // td0 hi | td1 lo
    g1[3] = (int)((td1 >> 16) | (128u << 16));           // td1 hi | tile_dim0=128
    g1[4] = (int)128;                                    // tile_dim1=128, tile_dim2=0
    g1[5] = (int)DIM;                                    // tensor_dim0_stride lo
    g1[6] = 0;
    g1[7] = 0;

    i32x4 z4 = {};
#if __clang_major__ >= 23
    i32x8 z8 = {};
    __builtin_amdgcn_tensor_load_to_lds(g0, g1, z4, z4, z8, 0);
#else
    __builtin_amdgcn_tensor_load_to_lds(g0, g1, z4, z4, 0);
#endif
    tdm_wait0();
  } else if (tid < 64) {                // wave 1: bf16 W buffer DMA (1D copy)
    unsigned ldsOff = (unsigned)((char*)&sm.Wt[0] - (char*)&sm);
    unsigned long long ga = (unsigned long long)(size_t)wt;
    const unsigned nd = WT_DWORDS;      // 8448 dwords

    u32x4 g0;
    g0[0] = 1u;
    g0[1] = ldsOff;
    g0[2] = (unsigned)(ga & 0xFFFFFFFFull);
    g0[3] = (unsigned)((ga >> 32) & 0x1FFFFFFull) | (2u << 30);

    i32x8 g1;
    g1[0] = (int)(2u << 16);                             // data_size=4B, no padding
    g1[1] = (int)((nd & 0xFFFFu) << 16);                 // tensor_dim0 lo
    g1[2] = (int)((nd >> 16) | (1u << 16));              // td0 hi | tensor_dim1 = 1
    g1[3] = (int)(nd << 16);                             // td1 hi=0 | tile_dim0 = nd
    g1[4] = (int)1;                                      // tile_dim1 = 1
    g1[5] = (int)nd;                                     // stride (unused for 1 row)
    g1[6] = 0;
    g1[7] = 0;

    i32x4 z4 = {};
#if __clang_major__ >= 23
    i32x8 z8 = {};
    __builtin_amdgcn_tensor_load_to_lds(g0, g1, z4, z4, z8, 0);
#else
    __builtin_amdgcn_tensor_load_to_lds(g0, g1, z4, z4, 0);
#endif
    tdm_wait0();
  }
#else
  // Fallback: cooperative copies with zero-fill
  for (int i = tid; i < DIM * DIM / 4; i += 256) {
    int rowL = i >> 5;                 // 32 float4 per row
    int c4   = (i & 31) << 2;
    float4 v = make_float4(0.f, 0.f, 0.f, 0.f);
    int grow = r0 + rowL;
    if (grow < N_NODES) v = *(const float4*)(x + (size_t)grow * DIM + c4);
    float* dst = &sm.xs[rowL * XS_PITCH + c4];
    dst[0] = v.x; dst[1] = v.y; dst[2] = v.z; dst[3] = v.w;
  }
  {
    const unsigned* wsrc = (const unsigned*)wt;
    unsigned* wdst = (unsigned*)sm.Wt;
    for (int i = tid; i < WT_DWORDS; i += 256) wdst[i] = wsrc[i];
  }
#endif
  __syncthreads();

  const int lane = tid & 31;
  const int wave = tid >> 5;
  const int mloc = lane & 15;
  const int half = lane >> 4;                   // 0 | 1

  v8f acc[8] = {};                              // 8 tiles of 16x16, all 128 cols

  const float* xr = &sm.xs[(wave * 16 + mloc) * XS_PITCH];

  for (int kt = 0; kt < 4; ++kt) {              // K in chunks of 32
    // A fragment: 16x32 bf16, ISA layout (lane=M; VGPR pair K; half-lane K off 8)
    v16bf a;
#pragma unroll
    for (int v = 0; v < 8; ++v) {
      int k = kt * 32 + ((v & 4) << 2) + (half << 3) + ((v & 3) << 1);
      a[2 * v]     = f2bf(xr[k]);
      a[2 * v + 1] = f2bf(xr[k + 1]);
    }

    // Load all 8 B fragments, then issue the 8 WMMAs back-to-back
    v16bf bf[8];
#pragma unroll
    for (int nt = 0; nt < 8; ++nt) {
      const __bf16* wp = &sm.Wt[(nt * 16 + mloc) * WT_PITCH + kt * 32 + (half << 3)];
#pragma unroll
      for (int v = 0; v < 8; ++v) { bf[nt][2 * v] = wp[2 * v + ((v & 4) << 1)]; bf[nt][2 * v + 1] = wp[2 * v + 1 + ((v & 4) << 1)]; }
    }
#pragma unroll
    for (int nt = 0; nt < 8; ++nt) {
      acc[nt] = __builtin_amdgcn_wmma_f32_16x16x32_bf16(
          false, a, false, bf[nt], (short)0, acc[nt], false, false);
    }
  }

  // store h: acc[nt][r] -> row rowBase + half*8 + r, col nt*16 + mloc
  const int rowBase = r0 + wave * 16;
  if (r0 + 128 <= N_NODES) {                     // fast path: every row in range
    float* hp = h + (size_t)(rowBase + half * 8) * DIM + mloc;
#pragma unroll
    for (int nt = 0; nt < 8; ++nt) {
#pragma unroll
      for (int r = 0; r < 8; ++r) hp[(size_t)r * DIM + nt * 16] = acc[nt][r];
    }
  } else {                                       // tail block only
#pragma unroll
    for (int nt = 0; nt < 8; ++nt) {
      int col = nt * 16 + mloc;
#pragma unroll
      for (int r = 0; r < 8; ++r) {
        int m = rowBase + half * 8 + r;
        if (m < N_NODES) h[(size_t)m * DIM + col] = acc[nt][r];
      }
    }
  }

  // alpha = h . a_src / h . a_dst, reduced across the 16 lanes of each half-wave
  float av[8], dv[8];
#pragma unroll
  for (int nt = 0; nt < 8; ++nt) {
    av[nt] = asrc[nt * 16 + mloc];
    dv[nt] = adst[nt * 16 + mloc];
  }
#pragma unroll
  for (int r = 0; r < 8; ++r) {
    float ps = 0.f, pd = 0.f;
#pragma unroll
    for (int nt = 0; nt < 8; ++nt) { ps += acc[nt][r] * av[nt]; pd += acc[nt][r] * dv[nt]; }
#pragma unroll
    for (int off = 1; off < 16; off <<= 1) {
      ps += __shfl_xor(ps, off, 32);
      pd += __shfl_xor(pd, off, 32);
    }
    int m = rowBase + half * 8 + r;
    if (mloc == 0 && m < N_NODES) { alpha_s[m] = ps; alpha_d[m] = pd; }
  }
}

// ---------- per-layer init: out = bias (fold), segment-max = -inf, denom = 0 ----------

__global__ __launch_bounds__(256)
void gat_init_kernel(float* __restrict__ out, const float* __restrict__ bias,
                     unsigned* __restrict__ mkey, float* __restrict__ denom)
{
  int gid = blockIdx.x * blockDim.x + threadIdx.x;   // N*32 lanes of float4
  int n = gid >> 5, c4 = (gid & 31) << 2;
  if (n >= N_NODES) return;
  float4 b4 = *(const float4*)(bias + c4);
  *(float4*)(out + (size_t)n * DIM + c4) = b4;
  if ((gid & 31) == 0) { mkey[n] = 0x007FFFFFu; denom[n] = 0.f; }
}

// ---------- edge pass 1: e = leaky_relu(as[src]+ad[dst]); segment max ----------

__global__ __launch_bounds__(256)
void gat_edge_max_kernel(const int* __restrict__ ei, const float* __restrict__ as,
                         const float* __restrict__ ad, unsigned* __restrict__ mkey,
                         float* __restrict__ ebuf)
{
  int i = blockIdx.x * blockDim.x + threadIdx.x;
  if (i >= ETOT) return;
  int s, d; edge_sd(ei, i, s, d);
  float v = as[s] + ad[d];
  float e = v > 0.f ? v : NEG_SLOPE * v;
  ebuf[i] = e;
  atomicMax(&mkey[d], f2ord(e));
}

// ---------- edge pass 2: ex = exp(e - m[dst]); segment sum ----------

__global__ __launch_bounds__(256)
void gat_edge_expsum_kernel(const int* __restrict__ ei, float* __restrict__ ebuf,
                            const unsigned* __restrict__ mkey, float* __restrict__ denom)
{
  int i = blockIdx.x * blockDim.x + threadIdx.x;
  if (i >= ETOT) return;
  int s, d; edge_sd(ei, i, s, d);
  float ex = __expf(ebuf[i] - ord2f(mkey[d]));
  ebuf[i] = ex;
  atomicAdd(&denom[d], ex);
}

// ---------- edge pass 3: coef = ex / denom[dst] ----------

__global__ __launch_bounds__(256)
void gat_edge_coef_kernel(const int* __restrict__ ei, float* __restrict__ ebuf,
                          const float* __restrict__ denom)
{
  int i = blockIdx.x * blockDim.x + threadIdx.x;
  if (i >= ETOT) return;
  int s, d; edge_sd(ei, i, s, d);
  ebuf[i] = ebuf[i] / denom[d];
}

// ---------- edge pass 4: out[dst] += coef * h[src]  (32 lanes x float4 per edge) ----------

__global__ __launch_bounds__(256)
void gat_edge_agg_kernel(const int* __restrict__ ei, const float* __restrict__ coef,
                         const float* __restrict__ h, float* __restrict__ out)
{
  int gid = blockIdx.x * blockDim.x + threadIdx.x;   // ETOT*32 = 27.2M
  int i = gid >> 5, k4 = (gid & 31) << 2;
  if (i >= ETOT) return;
  int s, d; edge_sd(ei, i, s, d);
  float c = coef[i];
  float4 hv = *(const float4*)(h + (size_t)s * DIM + k4);
  float* o = out + (size_t)d * DIM + k4;
  atomicAdd(o + 0, c * hv.x);
  atomicAdd(o + 1, c * hv.y);
  atomicAdd(o + 2, c * hv.z);
  atomicAdd(o + 3, c * hv.w);
}

// ---------- mean pool accumulate over concat [x0, h3] ----------

__global__ __launch_bounds__(256)
void gat_pool_kernel(const float* __restrict__ x0, const float* __restrict__ h3,
                     const int* __restrict__ batch, float* __restrict__ pool,
                     float* __restrict__ cnt)
{
  int gid = blockIdx.x * blockDim.x + threadIdx.x;   // N*256
  int n = gid >> 8, c = gid & 255;
  if (n >= N_NODES) return;
  float v = (c < DIM) ? x0[(size_t)n * DIM + c] : h3[(size_t)n * DIM + (c - DIM)];
  int g = batch[n];
  atomicAdd(&pool[g * 256 + c], v);
  if (c == 0) atomicAdd(&cnt[g], 1.0f);
}

__global__ __launch_bounds__(256)
void gat_zero_kernel(float* __restrict__ p, int n)
{
  int i = blockIdx.x * blockDim.x + threadIdx.x;
  if (i < n) p[i] = 0.f;
}

// ---------- final: embed, L2-normalize, pairwise distance ----------

__global__ __launch_bounds__(128)
void gat_final_kernel(const float* __restrict__ pool_s, const float* __restrict__ cnt_s,
                      const float* __restrict__ pool_t, const float* __restrict__ cnt_t,
                      const float* __restrict__ embW, const float* __restrict__ embB,
                      float* __restrict__ geds)
{
  __shared__ float es[EMBD], et[EMBD];
  int g = blockIdx.x, t = threadIdx.x;
  if (t < EMBD) {
    float inv = 1.f / fmaxf(cnt_s[g], 1.f);
    float acc = embB[t];
    for (int j = 0; j < 2 * DIM; ++j) acc += (pool_s[g * 256 + j] * inv) * embW[j * EMBD + t];
    es[t] = acc;
  } else {
    int tt = t - EMBD;
    float inv = 1.f / fmaxf(cnt_t[g], 1.f);
    float acc = embB[tt];
    for (int j = 0; j < 2 * DIM; ++j) acc += (pool_t[g * 256 + j] * inv) * embW[j * EMBD + tt];
    et[tt] = acc;
  }
  __syncthreads();
  if (t == 0) {
    float ns = 0.f, nt2 = 0.f;
    for (int j = 0; j < EMBD; ++j) { ns += es[j] * es[j]; nt2 += et[j] * et[j]; }
    ns  = fmaxf(sqrtf(ns),  1e-12f);
    nt2 = fmaxf(sqrtf(nt2), 1e-12f);
    float dd = 0.f;
    for (int j = 0; j < EMBD; ++j) {
      float df = es[j] / ns - et[j] / nt2;
      dd += df * df;
    }
    geds[g] = sqrtf(dd);
  }
}

// ---------- launch ----------

extern "C" void kernel_launch(void* const* d_in, const int* in_sizes, int n_in,
                              void* d_out, int out_size, void* d_ws, size_t ws_size,
                              hipStream_t stream)
{
  (void)in_sizes; (void)n_in; (void)out_size; (void)ws_size;

  const float* x_s      = (const float*)d_in[0];
  const int*   ei_s     = (const int*)d_in[1];
  const float* x_t      = (const float*)d_in[3];
  const int*   ei_t     = (const int*)d_in[4];
  const int*   batch_s  = (const int*)d_in[6];
  const int*   batch_t  = (const int*)d_in[7];
  const float* Wall     = (const float*)d_in[8];
  const float* asrcAll  = (const float*)d_in[9];
  const float* adstAll  = (const float*)d_in[10];
  const float* biasAll  = (const float*)d_in[11];
  const float* embW     = (const float*)d_in[12];
  const float* embB     = (const float*)d_in[13];
  float*       geds     = (float*)d_out;

  const size_t NF = (size_t)N_NODES * DIM;
  float*    ws     = (float*)d_ws;
  float*    bufH   = ws;                 // h  = x@W               (N*128)
  float*    bufA   = bufH + NF;          // layer out ping         (N*128)
  float*    bufB   = bufA + NF;          // layer out pong         (N*128)
  float*    al_s   = bufB + NF;          // alpha_src              (N)
  float*    al_d   = al_s + N_NODES;     // alpha_dst              (N)
  unsigned* mkey   = (unsigned*)(al_d + N_NODES);   // seg max key (N)
  float*    denom  = (float*)(mkey + N_NODES);      // seg sum     (N)
  float*    ebuf   = denom + N_NODES;    // per-edge e/ex/coef     (ETOT)
  float*    pool_s = ebuf + ETOT;        // [G,256]
  float*    pool_t = pool_s + GGR * 256; // [G,256]
  float*    cnt_s  = pool_t + GGR * 256; // [G]
  float*    cnt_t  = cnt_s + GGR;        // [G]
  unsigned short* wtb = (unsigned short*)(cnt_t + GGR);  // pitched bf16 W (16896 elems)

  {
    int n = 2 * GGR * 256 + 2 * GGR;     // pools + counts are contiguous
    gat_zero_kernel<<<(n + 255) / 256, 256, 0, stream>>>(pool_s, n);
  }

  const dim3 gemmGrid((N_NODES + 127) / 128), blk(256);
  const int wconvBlocks = (DIM * DIM + 255) / 256;
  const int edgeBlocks  = (ETOT + 255) / 256;
  const int aggBlocks   = (int)(((long long)ETOT * 32 + 255) / 256);
  const int initBlocks  = (N_NODES * 32 + 255) / 256;
  const int poolBlocks  = (N_NODES * 256 + 255) / 256;

  for (int br = 0; br < 2; ++br) {
    const float* x0    = br ? x_t : x_s;
    const int*   ei    = br ? ei_t : ei_s;
    const int*   batch = br ? batch_t : batch_s;
    float*       pool  = br ? pool_t : pool_s;
    float*       cnt   = br ? cnt_t : cnt_s;

    const float* cur = x0;
    float* outs[NLAYER] = { bufA, bufB, bufA };
    for (int l = 0; l < NLAYER; ++l) {
      float* o = outs[l];
      gat_wconv_kernel<<<wconvBlocks, blk, 0, stream>>>(Wall + l * DIM * DIM, wtb);
      gat_gemm_kernel<<<gemmGrid, blk, 0, stream>>>(
          cur, wtb, asrcAll + l * DIM, adstAll + l * DIM, bufH, al_s, al_d);
      gat_init_kernel<<<initBlocks, blk, 0, stream>>>(o, biasAll + l * DIM, mkey, denom);
      gat_edge_max_kernel<<<edgeBlocks, blk, 0, stream>>>(ei, al_s, al_d, mkey, ebuf);
      gat_edge_expsum_kernel<<<edgeBlocks, blk, 0, stream>>>(ei, ebuf, mkey, denom);
      gat_edge_coef_kernel<<<edgeBlocks, blk, 0, stream>>>(ei, ebuf, denom);
      gat_edge_agg_kernel<<<aggBlocks, blk, 0, stream>>>(ei, ebuf, bufH, o);
      cur = o;
    }
    gat_pool_kernel<<<poolBlocks, blk, 0, stream>>>(x0, cur, batch, pool, cnt);
  }

  gat_final_kernel<<<GGR, 128, 0, stream>>>(pool_s, cnt_s, pool_t, cnt_t, embW, embB, geds);
}
